// GNNCritic_42700564857465
// MI455X (gfx1250) — matrix operand order repaired and checked by reading, not compiled, not run on hardware
//
#include <hip/hip_runtime.h>
#include <hip/hip_bf16.h>
#include <math.h>

typedef __attribute__((ext_vector_type(16))) __bf16 v16bf;
typedef __attribute__((ext_vector_type(8)))  __bf16 bf16x8;
typedef __attribute__((ext_vector_type(4)))  __bf16 bf16x4;
typedef __attribute__((ext_vector_type(8)))  float  v8f;

#define DEV static __device__ __forceinline__

#define WMMA_BF16(a, b, c) \
  __builtin_amdgcn_wmma_f32_16x16x32_bf16(false, (a), false, (b), (short)0, (c), false, false)

// ---------------- fragment loaders (wave32, ISA 7.12.2 layouts) ----------------
// A fragment: lane row = lane&15; K base kb = k0 + (lane&16 ? 8 : 0);
// a[i] = A[row][kb+i] (i<8), a[8+i] = A[row][kb+16+i]  -> two 16B vector loads.

DEV v16bf load_a_bf16v(const __bf16* __restrict__ A, int lda, int row0, int k0, int lane) {
  int row = row0 + (lane & 15);
  int kb  = k0 + ((lane & 16) ? 8 : 0);
  const __bf16* base = A + (long)row * lda + kb;
  bf16x8 lo = *(const bf16x8*)base;
  bf16x8 hi = *(const bf16x8*)(base + 16);
  v16bf a;
#pragma unroll
  for (int i = 0; i < 8; ++i) { a[i] = lo[i]; a[8 + i] = hi[i]; }
  return a;
}

// A fragment from fp32 source (vector loads + convert), optional 1/max(deg,1) row scale,
// zero-pad of the upper K chunk when kmax is exceeded (for K=16 layers).
DEV v16bf load_a_f32v(const float* __restrict__ A, int lda, int row0, int k0, int kmax,
                      const float* __restrict__ deg, int lane) {
  int row = row0 + (lane & 15);
  int kb  = k0 + ((lane & 16) ? 8 : 0);
  float s = deg ? (1.0f / fmaxf(deg[row], 1.0f)) : 1.0f;
  const float* base = A + (long)row * lda + kb;
  float4 u0 = *(const float4*)(base);
  float4 u1 = *(const float4*)(base + 4);
  v16bf a;
  a[0] = (__bf16)(u0.x * s); a[1] = (__bf16)(u0.y * s);
  a[2] = (__bf16)(u0.z * s); a[3] = (__bf16)(u0.w * s);
  a[4] = (__bf16)(u1.x * s); a[5] = (__bf16)(u1.y * s);
  a[6] = (__bf16)(u1.z * s); a[7] = (__bf16)(u1.w * s);
  if (kb + 24 <= kmax) {
    float4 u2 = *(const float4*)(base + 16);
    float4 u3 = *(const float4*)(base + 20);
    a[8]  = (__bf16)(u2.x * s); a[9]  = (__bf16)(u2.y * s);
    a[10] = (__bf16)(u2.z * s); a[11] = (__bf16)(u2.w * s);
    a[12] = (__bf16)(u3.x * s); a[13] = (__bf16)(u3.y * s);
    a[14] = (__bf16)(u3.z * s); a[15] = (__bf16)(u3.w * s);
  } else {
#pragma unroll
    for (int i = 8; i < 16; ++i) a[i] = (__bf16)0.0f;
  }
  return a;
}

// B fragment from pre-packed weights: frag = kt*NT + nt, per-lane 16 contiguous bf16.
DEV v16bf load_b_packed(const __bf16* __restrict__ P, int frag, int lane) {
  const bf16x8* p = (const bf16x8*)(P + (((long)frag * 32 + lane) << 4));
  bf16x8 lo = p[0], hi = p[1];
  v16bf b;
#pragma unroll
  for (int i = 0; i < 8; ++i) { b[i] = lo[i]; b[8 + i] = hi[i]; }
  return b;
}

// Epilogue: wave owns a 16x64 strip (4 accumulators). Stage activated bf16 tile in
// wave-private LDS (C layout writes), then read back row-major and store 4x b128/lane.
DEV void epilogue_strip(__bf16* __restrict__ D, int ldd, int row0, int col0,
                        v8f c0, v8f c1, v8f c2, v8f c3,
                        const float* __restrict__ bias, int bcol0, int act, int lane,
                        __bf16* __restrict__ lds) {
  v8f cs[4] = {c0, c1, c2, c3};
  int col = lane & 15;
  int rb  = (lane & 16) ? 8 : 0;
#pragma unroll
  for (int nt = 0; nt < 4; ++nt) {
    float bv = bias[bcol0 + nt * 16 + col];
#pragma unroll
    for (int j = 0; j < 8; ++j) {
      float v = cs[nt][j] + bv;
      if (act == 1) v = fmaxf(v, 0.0f);
      else if (act == 2) v = tanhf(v);
      lds[(rb + j) * 64 + nt * 16 + col] = (__bf16)v;
    }
  }
  asm volatile("s_wait_dscnt 0x0" ::: "memory");  // LDS in-order per wave; be explicit
  int row  = lane >> 1;
  int half = (lane & 1) * 32;
  const bf16x8* s = (const bf16x8*)(lds + row * 64 + half);
  bf16x8* d = (bf16x8*)(D + (long)(row0 + row) * ldd + col0 + half);
  d[0] = s[0]; d[1] = s[1]; d[2] = s[2]; d[3] = s[3];
}

// ---------------- utility kernels ----------------

__global__ void k_zero(float* __restrict__ p, long n) {
  long i = (long)blockIdx.x * blockDim.x + threadIdx.x;
  long st = (long)gridDim.x * blockDim.x;
  for (; i < n; i += st) p[i] = 0.0f;
}

// Pack fp32 weights [K x N] into WMMA B-fragment layout: [kt][nt][lane][16] bf16.
// b[i] = W[kt*32 + (lane&16?8:0) + (i&7) + (i&8?16:0)][nt*16 + (lane&15)], zero-padded past K.
__global__ void k_packB(const float* __restrict__ W, __bf16* __restrict__ P, int K, int N) {
  int idx = blockIdx.x * blockDim.x + threadIdx.x;
  int NT = N >> 4;
  int Kt = (K + 31) >> 5;
  if (idx >= Kt * NT * 512) return;
  int i16  = idx & 15;
  int lane = (idx >> 4) & 31;
  int nt   = (idx >> 9) % NT;
  int kt   = idx / (NT << 9);
  int k    = (kt << 5) + ((lane & 16) ? 8 : 0) + (i16 & 7) + ((i16 & 8) ? 16 : 0);
  int col  = (nt << 4) + (lane & 15);
  P[idx] = (k < K) ? (__bf16)W[k * N + col] : (__bf16)0.0f;
}

__global__ void k_deg(const int* __restrict__ dst, float* __restrict__ deg, int E) {
  int e = blockIdx.x * blockDim.x + threadIdx.x;
  if (e < E) unsafeAtomicAdd(&deg[dst[e]], 1.0f);
}

// layer-0 neighbor sum: thread per (edge, 4-feature chunk), F=16
__global__ void k_scatter0(const float* __restrict__ obs, const int* __restrict__ src,
                           const int* __restrict__ dst, float* __restrict__ agg, int E) {
  int tid = blockIdx.x * blockDim.x + threadIdx.x;
  if (tid >= E * 4) return;
  int e = tid >> 2, c = (tid & 3) * 4;
  int s = src[e], d = dst[e];
  float4 v = *(const float4*)(obs + s * 16 + c);
  float* p = agg + d * 16 + c;
  unsafeAtomicAdd(p + 0, v.x);
  unsafeAtomicAdd(p + 1, v.y);
  unsafeAtomicAdd(p + 2, v.z);
  unsafeAtomicAdd(p + 3, v.w);
}

// layer-1 neighbor sum: thread per (edge, 4-feature chunk), H=128
__global__ void k_scatter1(const __bf16* __restrict__ h0, const int* __restrict__ src,
                           const int* __restrict__ dst, float* __restrict__ agg, int E) {
  int tid = blockIdx.x * blockDim.x + threadIdx.x;
  if (tid >= E * 32) return;
  int e = tid >> 5, c = (tid & 31) * 4;
  int s = src[e], d = dst[e];
  bf16x4 v = *(const bf16x4*)(h0 + s * 128 + c);
  float* p = agg + (long)d * 128 + c;
  unsafeAtomicAdd(p + 0, (float)v.x);
  unsafeAtomicAdd(p + 1, (float)v.y);
  unsafeAtomicAdd(p + 2, (float)v.z);
  unsafeAtomicAdd(p + 3, (float)v.w);
}

// ---------------- GNN layer 0: K=16 (padded to 32) ----------------
// grid = N/64, 8 waves: wave w -> mtile (w>>1), half (w&1): self cols 0..63 / neigh 64..127
__global__ void k_gnn0(const float* __restrict__ obs, const float* __restrict__ agg,
                       const float* __restrict__ deg,
                       const __bf16* __restrict__ pself, const __bf16* __restrict__ pneigh,
                       const float* __restrict__ bself, const float* __restrict__ bneigh,
                       __bf16* __restrict__ h0) {
  __shared__ __bf16 smem[8 * 16 * 64];
  int w = threadIdx.x >> 5, lane = threadIdx.x & 31;
  int row0 = blockIdx.x * 64 + (w >> 1) * 16;
  int half = w & 1;
  const __bf16* P = half ? pneigh : pself;
  v8f c0 = {}, c1 = {}, c2 = {}, c3 = {};
  v16bf a = half ? load_a_f32v(agg, 16, row0, 0, 16, deg, lane)
                 : load_a_f32v(obs, 16, row0, 0, 16, nullptr, lane);
  c0 = WMMA_BF16(a, load_b_packed(P, 0, lane), c0);
  c1 = WMMA_BF16(a, load_b_packed(P, 1, lane), c1);
  c2 = WMMA_BF16(a, load_b_packed(P, 2, lane), c2);
  c3 = WMMA_BF16(a, load_b_packed(P, 3, lane), c3);
  epilogue_strip(h0, 128, row0, half * 64, c0, c1, c2, c3,
                 half ? bneigh : bself, 0, 1, lane, smem + w * 1024);
}

// ---------------- GNN layer 1: K=128 ----------------
__global__ void k_gnn1(const __bf16* __restrict__ h0, const float* __restrict__ agg,
                       const float* __restrict__ deg,
                       const __bf16* __restrict__ pself, const __bf16* __restrict__ pneigh,
                       const float* __restrict__ bself, const float* __restrict__ bneigh,
                       __bf16* __restrict__ h1) {
  __shared__ __bf16 smem[8 * 16 * 64];
  int w = threadIdx.x >> 5, lane = threadIdx.x & 31;
  int row0 = blockIdx.x * 64 + (w >> 1) * 16;
  int half = w & 1;
  const __bf16* P = half ? pneigh : pself;
  v8f c0 = {}, c1 = {}, c2 = {}, c3 = {};
#pragma unroll
  for (int kt = 0; kt < 4; ++kt) {
    v16bf a = half ? load_a_f32v(agg, 128, row0, kt * 32, 128, deg, lane)
                   : load_a_bf16v(h0, 128, row0, kt * 32, lane);
    c0 = WMMA_BF16(a, load_b_packed(P, kt * 4 + 0, lane), c0);
    c1 = WMMA_BF16(a, load_b_packed(P, kt * 4 + 1, lane), c1);
    c2 = WMMA_BF16(a, load_b_packed(P, kt * 4 + 2, lane), c2);
    c3 = WMMA_BF16(a, load_b_packed(P, kt * 4 + 3, lane), c3);
  }
  epilogue_strip(h1, 128, row0, half * 64, c0, c1, c2, c3,
                 half ? bneigh : bself, 0, 1, lane, smem + w * 1024);
}

// ---------------- MLP GEMM: D = tanh(A[R x K] @ W[K x 256] + b) ----------------
// block = 8 waves covering 2 mtiles x 4 col-groups (16x64 strip per wave)
__global__ void k_mlp(const __bf16* __restrict__ A, const __bf16* __restrict__ W,
                      const float* __restrict__ bias, __bf16* __restrict__ D, int K) {
  __shared__ __bf16 smem[8 * 16 * 64];
  int w = threadIdx.x >> 5, lane = threadIdx.x & 31;
  int row0 = (blockIdx.x * 2 + (w >> 2)) * 16;
  int cg = w & 3;                       // col group: 64 cols each
  v8f c0 = {}, c1 = {}, c2 = {}, c3 = {};
  for (int kt = 0; kt < (K >> 5); ++kt) {
    v16bf a = load_a_bf16v(A, K, row0, kt * 32, lane);
    int f = kt * 16 + cg * 4;
    c0 = WMMA_BF16(a, load_b_packed(W, f + 0, lane), c0);
    c1 = WMMA_BF16(a, load_b_packed(W, f + 1, lane), c1);
    c2 = WMMA_BF16(a, load_b_packed(W, f + 2, lane), c2);
    c3 = WMMA_BF16(a, load_b_packed(W, f + 3, lane), c3);
  }
  epilogue_strip(D, 256, row0, cg * 64, c0, c1, c2, c3, bias, cg * 64, 2, lane,
                 smem + w * 1024);
}

// ---------------- final projection: out[r] = x1[r] . w2 + b2 ----------------
__global__ void k_mlp2(const __bf16* __restrict__ X, const float* __restrict__ w2,
                       const float* __restrict__ b2, float* __restrict__ out, int R) {
  int w = threadIdx.x >> 5, lane = threadIdx.x & 31;
  int r = blockIdx.x * 8 + w;
  if (r >= R) return;
  float s = 0.0f;
#pragma unroll
  for (int i = 0; i < 8; ++i) {
    int k = lane * 8 + i;
    s += (float)X[r * 256 + k] * w2[k];
  }
#pragma unroll
  for (int off = 16; off; off >>= 1) s += __shfl_xor(s, off, 32);
  if (lane == 0) out[r] = s + b2[0];
}

// ---------------- launch ----------------

extern "C" void kernel_launch(void* const* d_in, const int* in_sizes, int n_in,
                              void* d_out, int out_size, void* d_ws, size_t ws_size,
                              hipStream_t stream) {
  const float* obs = (const float*)d_in[0];
  const int*  edge = (const int*)d_in[1];
  const float* wS0 = (const float*)d_in[2];  const float* bS0 = (const float*)d_in[3];
  const float* wN0 = (const float*)d_in[4];  const float* bN0 = (const float*)d_in[5];
  const float* wS1 = (const float*)d_in[6];  const float* bS1 = (const float*)d_in[7];
  const float* wN1 = (const float*)d_in[8];  const float* bN1 = (const float*)d_in[9];
  const float* mW0 = (const float*)d_in[10]; const float* mB0 = (const float*)d_in[11];
  const float* mW1 = (const float*)d_in[12]; const float* mB1 = (const float*)d_in[13];
  const float* mW2 = (const float*)d_in[14]; const float* mB2 = (const float*)d_in[15];
  float* out = (float*)d_out;

  const int N = in_sizes[0] / 16;   // 131072 nodes
  const int E = in_sizes[1] / 2;    // 1048576 edges
  const int R = N / 8;              // 16384 MLP rows
  const int* src = edge;
  const int* dst = edge + E;

  char* ws = (char*)d_ws;
  size_t off = 0;
  auto alloc = [&](size_t bytes) -> void* {
    void* p = ws + off;
    off = (off + bytes + 255) & ~(size_t)255;
    return p;
  };
  float*  deg  = (float*)alloc((size_t)N * 4);
  float*  agg  = (float*)alloc((size_t)N * 128 * 4);   // agg0 (Nx16) then agg1 (Nx128)
  __bf16* h0   = (__bf16*)alloc((size_t)N * 128 * 2);
  __bf16* h1   = (__bf16*)alloc((size_t)N * 128 * 2);
  __bf16* pS0  = (__bf16*)alloc(32 * 64 * 2);          // K padded to 32
  __bf16* pN0  = (__bf16*)alloc(32 * 64 * 2);
  __bf16* pS1  = (__bf16*)alloc(128 * 64 * 2);
  __bf16* pN1  = (__bf16*)alloc(128 * 64 * 2);
  __bf16* pW0  = (__bf16*)alloc(1024 * 256 * 2);
  __bf16* pW1  = (__bf16*)alloc(256 * 256 * 2);
  // x0/x1 alias the aggregation buffer (dead after k_gnn1)
  __bf16* x0 = (__bf16*)agg;
  __bf16* x1 = (__bf16*)agg + (size_t)R * 256;

  // weight packing fp32 -> bf16 fragment layout
  k_packB<<<8, 256, 0, stream>>>(wS0, pS0, 16, 64);
  k_packB<<<8, 256, 0, stream>>>(wN0, pN0, 16, 64);
  k_packB<<<32, 256, 0, stream>>>(wS1, pS1, 128, 64);
  k_packB<<<32, 256, 0, stream>>>(wN1, pN1, 128, 64);
  k_packB<<<1024, 256, 0, stream>>>(mW0, pW0, 1024, 256);
  k_packB<<<256, 256, 0, stream>>>(mW1, pW1, 256, 256);

  // degrees + layer-0 aggregation
  k_zero<<<1024, 256, 0, stream>>>(deg, (long)N);
  k_zero<<<4096, 256, 0, stream>>>(agg, (long)N * 16);
  k_deg<<<(E + 255) / 256, 256, 0, stream>>>(dst, deg, E);
  k_scatter0<<<E * 4 / 256, 256, 0, stream>>>(obs, src, dst, agg, E);
  k_gnn0<<<N / 64, 256, 0, stream>>>(obs, agg, deg, pS0, pN0, bS0, bN0, h0);

  // layer-1 aggregation + GEMM
  k_zero<<<16384, 256, 0, stream>>>(agg, (long)N * 128);
  k_scatter1<<<E * 32 / 256, 256, 0, stream>>>(h0, src, dst, agg, E);
  k_gnn1<<<N / 64, 256, 0, stream>>>(h0, agg, deg, pS1, pN1, bS1, bN1, h1);

  // MLP: h1 viewed as [R x 1024] (phases contiguous per node group)
  k_mlp<<<R / 32, 256, 0, stream>>>(h1, pW0, mB0, x0, 1024);
  k_mlp<<<R / 32, 256, 0, stream>>>(x0, pW1, mB1, x1, 256);
  k_mlp2<<<R / 8, 256, 0, stream>>>(x1, mW2, mB2, out, R);
}
